// LSTMEDModule_31353261260841
// MI455X (gfx1250) — matrix operand order repaired and checked by reading, not compiled, not run on hardware
//
#include <hip/hip_runtime.h>
#include <hip/hip_bf16.h>
#include <math.h>

// ---------------- problem constants ----------------
#define T_SEQ 1024
#define B_SZ  128
#define F_SZ  64
#define H_SZ  512
#define G4    2048          // 4*H
#define NWG   32            // one WG per 16-wide hidden column tile
#define TPB   256           // 8 waves of 32 -> one wave per batch row tile

typedef __attribute__((ext_vector_type(16))) __bf16 v16bf;
typedef __attribute__((ext_vector_type(8)))  float  v8f;

// ---------------- workspace layout (bytes, all 256-aligned) ----------------
#define SZ_WHH   (H_SZ*G4*2u)          // 2 MB   (bf16, transposed [H][4H])
#define SZ_WIH   (F_SZ*G4*2u)          // 256 KB (bf16, transposed [F][4H])
#define SZ_WOUT  (H_SZ*F_SZ*2u)        // 64 KB  (bf16, transposed [H][F])
#define OFF_WHH_E  0u
#define OFF_WIH_E  (OFF_WHH_E + SZ_WHH)
#define OFF_WHH_D  (OFF_WIH_E + SZ_WIH)
#define OFF_WIH_D  (OFF_WHH_D + SZ_WHH)
#define OFF_WOUT   (OFF_WIH_D + SZ_WIH)
#define OFF_BIAS_E (OFF_WOUT + SZ_WOUT)      // f32 [2048] = b_ih+b_hh
#define OFF_BIAS_D (OFF_BIAS_E + G4*4u)
#define OFF_BOUT   (OFF_BIAS_D + G4*4u)      // f32 [64] (padded to 256)
#define OFF_HBF    (OFF_BOUT + 256u)         // bf16 h, double buffered: 2*[128][512]
#define SZ_HBF     (2u*B_SZ*H_SZ*2u)
#define OFF_C      (OFF_HBF + SZ_HBF)        // f32 c [128][512]
#define SZ_C       (B_SZ*H_SZ*4u)
#define OFF_YBF    (OFF_C + SZ_C)            // bf16 y [128][64]
#define SZ_YBF     (B_SZ*F_SZ*2u)
#define OFF_BAR    (OFF_YBF + SZ_YBF)        // 2 x u32 barrier (count, generation)

// ---------------- helpers ----------------
__device__ __forceinline__ unsigned short f2bf(float x) {
    unsigned u = __float_as_uint(x);
    if ((u & 0x7fffffffu) > 0x7f800000u) return (unsigned short)0x7fc0u;  // NaN
    unsigned r = u + 0x7fffu + ((u >> 16) & 1u);                          // RNE
    return (unsigned short)(r >> 16);
}
__device__ __forceinline__ unsigned packbf2(float lo, float hi) {
    return (unsigned)f2bf(lo) | ((unsigned)f2bf(hi) << 16);
}
__device__ __forceinline__ float sigm(float x) { return 1.0f / (1.0f + __expf(-x)); }

union Frag { v16bf v; unsigned u[8]; };

// A fragment (16xK, 16-bit): lanes 0-15 -> M=lane; regs 0-3 pack K=khi*8+{0..7},
// regs 4-7 pack +16.  src row-major bf16, kbase multiple of 32.
__device__ __forceinline__ void loadA_bf(Frag& f, const unsigned short* __restrict__ base,
                                         int ld, int kbase, int lane) {
    int m = lane & 15, khi = lane >> 4;
    const unsigned* p = (const unsigned*)(base + (size_t)m * ld + kbase + khi * 8);
    f.u[0] = p[0]; f.u[1] = p[1]; f.u[2] = p[2]; f.u[3] = p[3];
    const unsigned* q = p + 8;   // +16 bf16 elements
    f.u[4] = q[0]; f.u[5] = q[1]; f.u[6] = q[2]; f.u[7] = q[3];
}

// Same A layout but f32 source converted on the fly (encoder input rows).
__device__ __forceinline__ void loadA_f32(Frag& f, const float* __restrict__ base,
                                          size_t ld, int kbase, int lane) {
    int m = lane & 15, khi = lane >> 4;
    const float* p = base + (size_t)m * ld + kbase + khi * 8;
#pragma unroll
    for (int v = 0; v < 4; ++v) f.u[v] = packbf2(p[2 * v], p[2 * v + 1]);
    p += 16;
#pragma unroll
    for (int v = 0; v < 4; ++v) f.u[4 + v] = packbf2(p[2 * v], p[2 * v + 1]);
}

// B fragment from an LDS-staged 32x16 bf16 tile: N = lane%16, K in A's table order.
__device__ __forceinline__ void loadB_lds(Frag& f, const unsigned short* tile, int lane) {
    int n = lane & 15, h = lane >> 4;
#pragma unroll
    for (int v = 0; v < 8; ++v) {
        int k = (v < 4) ? (8 * h + 2 * v) : (16 + 8 * h + 2 * (v - 4));
        f.u[v] = (unsigned)tile[k * 16 + n] | ((unsigned)tile[(k + 1) * 16 + n] << 16);
    }
}

// B fragment straight from global transposed weights [K][ld] bf16.
__device__ __forceinline__ void loadB_gl(Frag& f, const unsigned short* __restrict__ W,
                                         int ld, int kbase, int col0, int lane) {
    int n = lane & 15, h = lane >> 4;
#pragma unroll
    for (int v = 0; v < 8; ++v) {
        int k = kbase + ((v < 4) ? (8 * h + 2 * v) : (16 + 8 * h + 2 * (v - 4)));
        f.u[v] = (unsigned)W[(size_t)k * ld + col0 + n] | ((unsigned)W[(size_t)(k + 1) * ld + col0 + n] << 16);
    }
}

__device__ __forceinline__ v8f wmma_bf16(const Frag& a, const Frag& b, v8f c) {
    return __builtin_amdgcn_wmma_f32_16x16x32_bf16(false, a.v, false, b.v,
                                                   (short)0, c, false, false);
}

// Generation-count barrier across the fully-resident grid.
__device__ __forceinline__ void grid_barrier(unsigned* __restrict__ bar) {
    __threadfence();
    __syncthreads();
    if (threadIdx.x == 0) {
        unsigned* cnt = bar;
        unsigned* gen = bar + 1;
        unsigned g = __hip_atomic_load(gen, __ATOMIC_RELAXED, __HIP_MEMORY_SCOPE_AGENT);
        unsigned a = __hip_atomic_fetch_add(cnt, 1u, __ATOMIC_ACQ_REL, __HIP_MEMORY_SCOPE_AGENT);
        if (a == (unsigned)(NWG - 1)) {
            __hip_atomic_store(cnt, 0u, __ATOMIC_RELAXED, __HIP_MEMORY_SCOPE_AGENT);
            __hip_atomic_fetch_add(gen, 1u, __ATOMIC_RELEASE, __HIP_MEMORY_SCOPE_AGENT);
        } else {
            while (__hip_atomic_load(gen, __ATOMIC_ACQUIRE, __HIP_MEMORY_SCOPE_AGENT) == g)
                __builtin_amdgcn_s_sleep(1);
        }
    }
    __syncthreads();
}

// One LSTM cell step.  WG = hidden column tile ct (all 4 gates), wave = batch
// row tile rt.  B tiles (shared by all 8 waves) are double-buffered through
// LDS with an explicitly two-phase (buf0/buf1) pipeline so the compiler needs
// no register rotation for the loop-carried accumulators/fragments.
template <bool ENC>
__device__ __forceinline__ void gate_step(
    const unsigned short* __restrict__ Whh,   // [512][2048] bf16
    const unsigned short* __restrict__ Wih,   // [64][2048]  bf16
    const float* __restrict__ bias,           // [2048] f32 (b_ih+b_hh)
    const unsigned short* __restrict__ hread, // [128][512] bf16
    unsigned short* __restrict__ hwrite,      // [128][512] bf16
    float* __restrict__ c,                    // [128][512] f32
    const float* __restrict__ xf, size_t xld, // ENC: f32 input rows (step t)
    const unsigned short* __restrict__ xb,    // !ENC: bf16 y rows [128][64]
    unsigned short (*sB)[4][32][16],          // LDS [2][gate][K][N]
    int rt, int ct, int lane) {
    const int rowbase = rt * 16;
    const unsigned short* hrow = hread + (size_t)rowbase * H_SZ;

    v8f acc0 = (v8f){0, 0, 0, 0, 0, 0, 0, 0};
    v8f acc1 = (v8f){0, 0, 0, 0, 0, 0, 0, 0};
    v8f acc2 = (v8f){0, 0, 0, 0, 0, 0, 0, 0};
    v8f acc3 = (v8f){0, 0, 0, 0, 0, 0, 0, 0};

    // cooperative staging geometry: 256 threads x 16B = one 4 KB block (4 tiles)
    const int st_t = threadIdx.x >> 6;          // gate tile 0..3
    const int st_r = (threadIdx.x >> 1) & 31;   // K row 0..31
    const int st_h = threadIdx.x & 1;           // 16B half of 32B row
    const int st_col = ct * 16 + st_h * 8;      // ushort column in W

    Frag a0, a1, b0, b1, b2, b3;

#define STAGE_LD(W_, kb_) (*(const uint4*)((W_) + (size_t)((kb_) + st_r) * G4 + st_t * H_SZ + st_col))
#define STAGE_ST(buf_, v_) (*(uint4*)&sB[(buf_)][st_t][st_r][st_h * 8] = (v_))
#define COMPUTE(buf_, afrag_)                                        \
    do {                                                             \
        loadB_lds(b0, &sB[(buf_)][0][0][0], lane);                   \
        loadB_lds(b1, &sB[(buf_)][1][0][0], lane);                   \
        loadB_lds(b2, &sB[(buf_)][2][0][0], lane);                   \
        loadB_lds(b3, &sB[(buf_)][3][0][0], lane);                   \
        acc0 = wmma_bf16((afrag_), b0, acc0);                        \
        acc1 = wmma_bf16((afrag_), b1, acc1);                        \
        acc2 = wmma_bf16((afrag_), b2, acc2);                        \
        acc3 = wmma_bf16((afrag_), b3, acc3);                        \
    } while (0)

    // ================= hidden GEMM: K = 512 (16 blocks, 8 x 2 phases) ========
    STAGE_ST(0, STAGE_LD(Whh, 0));
    loadA_bf(a0, hrow, H_SZ, 0, lane);
    __syncthreads();
#pragma unroll 1
    for (int kb = 0; kb < H_SZ; kb += 64) {
        // phase 0: compute buf0 (kb), prepare buf1 (kb+32)
        {
            uint4 s1 = STAGE_LD(Whh, kb + 32);
            loadA_bf(a1, hrow, H_SZ, kb + 32, lane);
            COMPUTE(0, a0);
            STAGE_ST(1, s1);
            __syncthreads();
        }
        // phase 1: compute buf1 (kb+32), prepare buf0 (kb+64)
        {
            const bool more = (kb + 64) < H_SZ;
            uint4 s0;
            if (more) {
                s0 = STAGE_LD(Whh, kb + 64);
                loadA_bf(a0, hrow, H_SZ, kb + 64, lane);
            }
            COMPUTE(1, a1);
            if (more) STAGE_ST(0, s0);
            __syncthreads();
        }
    }

    // ================= input GEMM: K = 64 (2 blocks, explicit) ===============
    STAGE_ST(0, STAGE_LD(Wih, 0));
    if (ENC) loadA_f32(a0, xf + (size_t)rowbase * xld, xld, 0, lane);
    else     loadA_bf(a0, xb + (size_t)rowbase * F_SZ, F_SZ, 0, lane);
    __syncthreads();
    {
        uint4 s1 = STAGE_LD(Wih, 32);
        if (ENC) loadA_f32(a1, xf + (size_t)rowbase * xld, xld, 32, lane);
        else     loadA_bf(a1, xb + (size_t)rowbase * F_SZ, F_SZ, 32, lane);
        COMPUTE(0, a0);
        STAGE_ST(1, s1);
        __syncthreads();
    }
    COMPUTE(1, a1);
    // no trailing sync needed: nothing touches sB until after the grid barrier

#undef STAGE_LD
#undef STAGE_ST
#undef COMPUTE

    // ---- fused elementwise LSTM cell, entirely in registers ----
    const int col = ct * 16 + (lane & 15);
    const float bi  = bias[0 * H_SZ + col];
    const float bf_ = bias[1 * H_SZ + col];
    const float bg  = bias[2 * H_SZ + col];
    const float bo  = bias[3 * H_SZ + col];
#pragma unroll
    for (int r = 0; r < 8; ++r) {
        const int brow = rowbase + r + ((lane >> 4) << 3);
        const size_t idx = (size_t)brow * H_SZ + col;
        const float iv = sigm(acc0[r] + bi);
        const float fv = sigm(acc1[r] + bf_);
        const float gv = tanhf(acc2[r] + bg);
        const float ov = sigm(acc3[r] + bo);
        const float cn = fv * c[idx] + iv * gv;
        c[idx] = cn;
        hwrite[idx] = f2bf(ov * tanhf(cn));
    }
}

// Decoder projection y = h @ Wout^T + b_out for one 16x16 tile (blocks 0..3).
// Software-pipelined: fragments for kb+32 are loaded before the WMMA on kb.
__device__ __forceinline__ void y_step(const unsigned short* __restrict__ Wout,
                                       const float* __restrict__ bout,
                                       const unsigned short* __restrict__ hread,
                                       unsigned short* __restrict__ ybf,
                                       float* __restrict__ out, int tout,
                                       int rt, int fct, int lane) {
    const int rowbase = rt * 16;
    v8f acc = (v8f){0, 0, 0, 0, 0, 0, 0, 0};
    Frag a0, a1, b0, b1;
    const unsigned short* hrow = hread + (size_t)rowbase * H_SZ;

    loadA_bf(a0, hrow, H_SZ, 0, lane);
    loadB_gl(b0, Wout, F_SZ, 0, fct * 16, lane);
#pragma unroll 1
    for (int kb = 0; kb < H_SZ; kb += 64) {
        loadA_bf(a1, hrow, H_SZ, kb + 32, lane);
        loadB_gl(b1, Wout, F_SZ, kb + 32, fct * 16, lane);
        acc = wmma_bf16(a0, b0, acc);
        if (kb + 64 < H_SZ) {
            loadA_bf(a0, hrow, H_SZ, kb + 64, lane);
            loadB_gl(b0, Wout, F_SZ, kb + 64, fct * 16, lane);
        }
        acc = wmma_bf16(a1, b1, acc);
    }
    const int f = fct * 16 + (lane & 15);
    const float bb = bout[f];
#pragma unroll
    for (int r = 0; r < 8; ++r) {
        const int brow = rowbase + r + ((lane >> 4) << 3);
        const float yv = acc[r] + bb;
        out[((size_t)brow * T_SEQ + tout) * F_SZ + f] = yv;
        ybf[brow * F_SZ + f] = f2bf(yv);
    }
}

// ---------------- prologue: weight quantize/transpose + state init ----------------
__global__ void lstm_init(const float* __restrict__ w_ih_e, const float* __restrict__ w_hh_e,
                          const float* __restrict__ b_ih_e, const float* __restrict__ b_hh_e,
                          const float* __restrict__ w_ih_d, const float* __restrict__ w_hh_d,
                          const float* __restrict__ b_ih_d, const float* __restrict__ b_hh_d,
                          const float* __restrict__ w_out, const float* __restrict__ b_out,
                          unsigned char* __restrict__ ws) {
    unsigned short* WhhE = (unsigned short*)(ws + OFF_WHH_E);
    unsigned short* WihE = (unsigned short*)(ws + OFF_WIH_E);
    unsigned short* WhhD = (unsigned short*)(ws + OFF_WHH_D);
    unsigned short* WihD = (unsigned short*)(ws + OFF_WIH_D);
    unsigned short* Wout = (unsigned short*)(ws + OFF_WOUT);
    float* biasE = (float*)(ws + OFF_BIAS_E);
    float* biasD = (float*)(ws + OFF_BIAS_D);
    float* bout  = (float*)(ws + OFF_BOUT);
    unsigned short* hbf = (unsigned short*)(ws + OFF_HBF);
    float* c = (float*)(ws + OFF_C);
    unsigned* bar = (unsigned*)(ws + OFF_BAR);

    const int tid = blockIdx.x * blockDim.x + threadIdx.x;
    const int nth = gridDim.x * blockDim.x;

    for (int i = tid; i < H_SZ * G4; i += nth) {          // [4H,H] -> [H,4H]
        int k = i / G4, j = i % G4;
        WhhE[i] = f2bf(w_hh_e[(size_t)j * H_SZ + k]);
        WhhD[i] = f2bf(w_hh_d[(size_t)j * H_SZ + k]);
    }
    for (int i = tid; i < F_SZ * G4; i += nth) {          // [4H,F] -> [F,4H]
        int k = i / G4, j = i % G4;
        WihE[i] = f2bf(w_ih_e[(size_t)j * F_SZ + k]);
        WihD[i] = f2bf(w_ih_d[(size_t)j * F_SZ + k]);
    }
    for (int i = tid; i < H_SZ * F_SZ; i += nth) {        // [F,H] -> [H,F]
        int k = i / F_SZ, j = i % F_SZ;
        Wout[i] = f2bf(w_out[(size_t)j * H_SZ + k]);
    }
    for (int i = tid; i < G4; i += nth) {
        biasE[i] = b_ih_e[i] + b_hh_e[i];
        biasD[i] = b_ih_d[i] + b_hh_d[i];
    }
    for (int i = tid; i < F_SZ; i += nth) bout[i] = b_out[i];
    for (int i = tid; i < 2 * B_SZ * H_SZ; i += nth) hbf[i] = 0;   // bf16 zero
    for (int i = tid; i < B_SZ * H_SZ; i += nth) c[i] = 0.0f;
    if (tid < 2) bar[tid] = 0u;                                    // fresh barrier each call
}

// ---------------- persistent encoder/decoder kernel ----------------
__global__ __launch_bounds__(TPB, 1) void lstm_main(const float* __restrict__ ts,
                                                    unsigned char* __restrict__ ws,
                                                    float* __restrict__ out) {
    __shared__ unsigned short sB[2][4][32][16];   // 8 KB double-buffered B tiles

    const int lane = threadIdx.x & 31;
    const int rt = threadIdx.x >> 5;     // wave -> batch row tile (0..7)
    const int ct = blockIdx.x;           // WG   -> hidden column tile (0..31)

    const unsigned short* WhhE = (const unsigned short*)(ws + OFF_WHH_E);
    const unsigned short* WihE = (const unsigned short*)(ws + OFF_WIH_E);
    const unsigned short* WhhD = (const unsigned short*)(ws + OFF_WHH_D);
    const unsigned short* WihD = (const unsigned short*)(ws + OFF_WIH_D);
    const unsigned short* Wout = (const unsigned short*)(ws + OFF_WOUT);
    const float* biasE = (const float*)(ws + OFF_BIAS_E);
    const float* biasD = (const float*)(ws + OFF_BIAS_D);
    const float* bout  = (const float*)(ws + OFF_BOUT);
    unsigned short* hbf = (unsigned short*)(ws + OFF_HBF);
    float* c = (float*)(ws + OFF_C);
    unsigned short* ybf = (unsigned short*)(ws + OFF_YBF);
    unsigned* bar = (unsigned*)(ws + OFF_BAR);

    const size_t BH = (size_t)B_SZ * H_SZ;
    int pb = 0;  // h read-buffer parity

    // ---- encoder: 1 grid barrier per step (h double-buffered) ----
    for (int t = 0; t < T_SEQ; ++t) {
        gate_step<true>(WhhE, WihE, biasE, hbf + (size_t)pb * BH, hbf + (size_t)(1 - pb) * BH,
                        c, ts + (size_t)t * F_SZ, (size_t)T_SEQ * F_SZ, nullptr,
                        sB, rt, ct, lane);
        grid_barrier(bar);
        pb ^= 1;
    }

    // decoder restarts with c0 = 0 (reference semantics)
    for (int i = threadIdx.x + blockIdx.x * TPB; i < B_SZ * H_SZ; i += NWG * TPB) c[i] = 0.0f;
    grid_barrier(bar);

    // ---- decoder: y projection, barrier, cell update, barrier ----
    for (int k = 0; k < T_SEQ; ++k) {
        if (blockIdx.x < 4)
            y_step(Wout, bout, hbf + (size_t)pb * BH, ybf, out, T_SEQ - 1 - k,
                   rt, blockIdx.x, lane);
        grid_barrier(bar);
        gate_step<false>(WhhD, WihD, biasD, hbf + (size_t)pb * BH, hbf + (size_t)(1 - pb) * BH,
                         c, nullptr, 0, ybf, sB, rt, ct, lane);
        grid_barrier(bar);
        pb ^= 1;
    }
}

extern "C" void kernel_launch(void* const* d_in, const int* in_sizes, int n_in,
                              void* d_out, int out_size, void* d_ws, size_t ws_size,
                              hipStream_t stream) {
    (void)in_sizes; (void)n_in; (void)out_size; (void)ws_size;
    const float* ts      = (const float*)d_in[0];
    const float* w_ih_e  = (const float*)d_in[1];
    const float* w_hh_e  = (const float*)d_in[2];
    const float* b_ih_e  = (const float*)d_in[3];
    const float* b_hh_e  = (const float*)d_in[4];
    const float* w_ih_d  = (const float*)d_in[5];
    const float* w_hh_d  = (const float*)d_in[6];
    const float* b_ih_d  = (const float*)d_in[7];
    const float* b_hh_d  = (const float*)d_in[8];
    const float* w_out   = (const float*)d_in[9];
    const float* b_out   = (const float*)d_in[10];
    unsigned char* ws = (unsigned char*)d_ws;
    float* out = (float*)d_out;

    lstm_init<<<256, 256, 0, stream>>>(w_ih_e, w_hh_e, b_ih_e, b_hh_e,
                                       w_ih_d, w_hh_d, b_ih_d, b_hh_d,
                                       w_out, b_out, ws);
    lstm_main<<<NWG, TPB, 0, stream>>>(ts, ws, out);
}
